// MPBlock_21809843929774
// MI455X (gfx1250) — matrix operand order repaired and checked by reading, not compiled, not run on hardware
//
#include <hip/hip_runtime.h>

// ---------------------------------------------------------------------------
// MPBlock on gfx1250: LayerNorm -> edge MLP (WMMA f16) -> scatter-add -> out GEMM
// D=128, H=256, N=50000, E=640000. All GEMMs on v_wmma_f32_16x16x32_f16.
// Weights staged in LDS (128KB) per block; fast v_rcp SiLU.
// ---------------------------------------------------------------------------

#define D    128
#define HID  256
#define NN   50000
#define NE   640000

typedef __attribute__((ext_vector_type(16))) _Float16 v16h;
typedef __attribute__((ext_vector_type(8)))  _Float16 v8h;
typedef __attribute__((ext_vector_type(8)))  float    v8f;

// Fast SiLU: x * rcp(1+exp(-x)) using v_rcp_f32 (1 ulp) instead of IEEE divide.
__device__ __forceinline__ float siluf(float x) {
    return x * __builtin_amdgcn_rcpf(1.0f + __expf(-x));
}

__device__ __forceinline__ v8f wmma_step(v16h a, v16h b, v8f c) {
    // D = A(16x32 f16) * B(32x16 f16) + C(16x16 f32)
    return __builtin_amdgcn_wmma_f32_16x16x32_f16(
        /*neg_a=*/false, a, /*neg_b=*/false, b,
        /*c_mod=*/(short)0, c, /*reuse_a=*/false, /*reuse_b=*/false);
}

// B operand: 8 f16 at p and 8 f16 at p+16 (lane supplies K=8*g..8*g+7 and 16+8*g..16+8*g+7 for col n).
// Works for both global and LDS pointers (addrspace inferred).
__device__ __forceinline__ v16h load_b16(const _Float16* p) {
    v8h lo = *(const v8h*)(p);
    v8h hi = *(const v8h*)(p + 16);
    v16h b;
#pragma unroll
    for (int i = 0; i < 8; ++i) { b[i] = lo[i]; b[i + 8] = hi[i]; }
    return b;
}

// Wave-local LDS sync: all 32 lanes run in lockstep and DS ops complete
// in-order per wave, so draining DScnt makes all lanes' stores visible.
__device__ __forceinline__ void wave_lds_sync() {
    asm volatile("s_wait_dscnt 0" ::: "memory");
}

// ---------------------------------------------------------------------------
// Kernel 1: h = LayerNorm(x); agg = 0. One wave (32 lanes) per node.
// ---------------------------------------------------------------------------
__global__ __launch_bounds__(256) void ln_zero_kernel(
    const float* __restrict__ x, const float* __restrict__ gamma,
    const float* __restrict__ beta, float* __restrict__ h,
    float* __restrict__ agg)
{
    int wave = threadIdx.x >> 5;
    int lane = threadIdx.x & 31;
    int node = blockIdx.x * 8 + wave;
    if (node >= NN) return;
    const float* xp = x + (size_t)node * D;
    float v[4], s = 0.f, s2 = 0.f;
#pragma unroll
    for (int i = 0; i < 4; ++i) {
        v[i] = xp[lane + 32 * i];
        s += v[i]; s2 += v[i] * v[i];
    }
#pragma unroll
    for (int off = 16; off > 0; off >>= 1) {
        s  += __shfl_xor(s,  off);
        s2 += __shfl_xor(s2, off);
    }
    float mu  = s * (1.0f / 128.0f);
    float var = s2 * (1.0f / 128.0f) - mu * mu;
    float inv = rsqrtf(var + 1e-5f);
    float* hp = h   + (size_t)node * D;
    float* ap = agg + (size_t)node * D;
#pragma unroll
    for (int i = 0; i < 4; ++i) {
        int k = lane + 32 * i;
        hp[k] = (v[i] - mu) * inv * gamma[k] + beta[k];
        ap[k] = 0.0f;
    }
}

// ---------------------------------------------------------------------------
// Kernel 2: transpose + f16-convert weights.
// W1[D][H] -> W1T[H][D], W2[H][D] -> W2T[D][H], Wt[D][D] -> WtT[D][D]
// W1T and W2T are contiguous (128 KB) so the edge kernel can bulk-copy them.
// ---------------------------------------------------------------------------
__global__ __launch_bounds__(256) void prep_weights_kernel(
    const float* __restrict__ W1, const float* __restrict__ W2,
    const float* __restrict__ Wt,
    _Float16* __restrict__ W1T, _Float16* __restrict__ W2T,
    _Float16* __restrict__ WtT)
{
    int i = blockIdx.x * 256 + threadIdx.x;
    if (i < D * HID) {
        int k = i / HID, n = i % HID;
        W1T[(size_t)n * D + k] = (_Float16)W1[i];
    }
    if (i < HID * D) {
        int k = i / D, d = i % D;
        W2T[(size_t)d * HID + k] = (_Float16)W2[i];
    }
    if (i < D * D) {
        int k = i / D, d = i % D;
        WtT[(size_t)d * D + k] = (_Float16)Wt[i];
    }
}

// ---------------------------------------------------------------------------
// Kernel 3: per-edge MLP + scatter-add. One wave per 16-edge tile.
//   s = e + h[c] + h[n];  t = silu(s) @ W1 + b1;  th = silu(t) @ W2 + b2
//   atomicAdd(agg[c], h[n] * th)
// LDS: W1T (64KB) + W2T (64KB) staged once per block + 4x8KB transpose tiles.
// ---------------------------------------------------------------------------
__global__ __launch_bounds__(128) void edge_mlp_kernel(
    const float* __restrict__ edge_emb,
    const int*   __restrict__ center,
    const int*   __restrict__ neighbor,
    const float* __restrict__ h,
    const _Float16* __restrict__ W1T,   // [HID][D], W2T follows contiguously
    const float* __restrict__ b1,
    const float* __restrict__ b2,
    float* __restrict__ agg)
{
    __shared__ _Float16 w1s[HID * D];          // 64 KB
    __shared__ _Float16 w2s[D * HID];          // 64 KB
    __shared__ _Float16 tstage[4][16 * HID];   // 4 x 8 KB

    // ---- Stage both weight matrices into LDS (128 KB, b128 copies) ----
    {
        const float4* src = (const float4*)W1T;     // 8192 x 16B total
        float4* dst = (float4*)w1s;                  // w1s,w2s contiguous? not
        // copy W1T -> w1s (4096 chunks) and W2T -> w2s (4096 chunks)
#pragma unroll 4
        for (int j = 0; j < 32; ++j)
            dst[threadIdx.x + 128 * j] = src[threadIdx.x + 128 * j];
        const float4* src2 = src + HID * D / 8;
        float4* dst2 = (float4*)w2s;
#pragma unroll 4
        for (int j = 0; j < 32; ++j)
            dst2[threadIdx.x + 128 * j] = src2[threadIdx.x + 128 * j];
    }
    __syncthreads();   // uniform: every thread reaches this

    int wave = threadIdx.x >> 5;
    int lane = threadIdx.x & 31;
    int g = lane >> 4;      // half-wave group
    int n = lane & 15;      // col for B/C layout; row for A layout
    int tile = blockIdx.x * 4 + wave;
    if (tile * 16 >= NE) return;
    int e0 = tile * 16;

    // ---- Build A chunks (silu(s), 16x128 f16, A-layout) in registers ----
    int arow = e0 + n;                     // this lane supplies row n of the tile
    const float* eb  = edge_emb + (size_t)arow * D;
    const float* hcp = h + (size_t)center[arow]   * D;
    const float* hnp = h + (size_t)neighbor[arow] * D;

    v16h A1[4];
#pragma unroll
    for (int c = 0; c < 4; ++c) {
        v16h a;
#pragma unroll
        for (int j = 0; j < 8; ++j) {
            int kb = 32 * c + ((j < 4) ? (8 * g + 2 * j)
                                       : (16 + 8 * g + 2 * (j - 4)));
            float2 ev = *(const float2*)(eb  + kb);
            float2 cv = *(const float2*)(hcp + kb);
            float2 nv = *(const float2*)(hnp + kb);
            a[2 * j]     = (_Float16)siluf(ev.x + cv.x + nv.x);
            a[2 * j + 1] = (_Float16)siluf(ev.y + cv.y + nv.y);
        }
        A1[c] = a;
    }

    // ---- Layer 1: 16x128 @ 128x256, silu, stage f16 to LDS ----
    _Float16* tbuf = tstage[wave];
    for (int t = 0; t < 16; ++t) {
        float bias = b1[16 * t + n];
        v8f acc;
#pragma unroll
        for (int r = 0; r < 8; ++r) acc[r] = bias;
#pragma unroll
        for (int c = 0; c < 4; ++c) {
            const _Float16* bp = w1s + (16 * t + n) * D + 32 * c + 8 * g;
            acc = wmma_step(A1[c], load_b16(bp), acc);
        }
#pragma unroll
        for (int r = 0; r < 8; ++r) {   // C layout: element r -> row 8*g+r, col 16*t+n
            tbuf[(8 * g + r) * HID + 16 * t + n] = (_Float16)siluf(acc[r]);
        }
    }
    wave_lds_sync();

    // ---- Reload staged tile in A-layout (16x256, 8 K-chunks) ----
    v16h A2[8];
#pragma unroll
    for (int c = 0; c < 8; ++c) {
        const _Float16* p = tbuf + n * HID + 32 * c + 8 * g;
        v8h lo = *(const v8h*)(p);        // K = 32c + 8g .. +7
        v8h hi = *(const v8h*)(p + 16);   // K = 32c + 16 + 8g .. +7
        v16h a;
#pragma unroll
        for (int i = 0; i < 8; ++i) { a[i] = lo[i]; a[i + 8] = hi[i]; }
        A2[c] = a;
    }

    // indices for the 8 rows this lane touches in C layout
    int rc[8], rn[8];
#pragma unroll
    for (int r = 0; r < 8; ++r) {
        int e = e0 + 8 * g + r;
        rc[r] = center[e];
        rn[r] = neighbor[e];
    }

    // ---- Layer 2: 16x256 @ 256x128, msg = h_n * theta, scatter-add ----
    for (int t2 = 0; t2 < 8; ++t2) {
        int d = 16 * t2 + n;
        float bias = b2[d];
        v8f acc;
#pragma unroll
        for (int r = 0; r < 8; ++r) acc[r] = bias;
#pragma unroll
        for (int c = 0; c < 8; ++c) {
            const _Float16* bp = w2s + d * HID + 32 * c + 8 * g;
            acc = wmma_step(A2[c], load_b16(bp), acc);
        }
#pragma unroll
        for (int r = 0; r < 8; ++r) {
            float hn = h[(size_t)rn[r] * D + d];
            atomicAdd(&agg[(size_t)rc[r] * D + d], acc[r] * hn);
        }
    }
}

// ---------------------------------------------------------------------------
// Kernel 4: out = silu(h + agg) @ Wt + bt. One wave per 16-node tile.
// WtT is 32KB -> stays WGP$-resident; load B from global.
// ---------------------------------------------------------------------------
__global__ __launch_bounds__(128) void node_out_kernel(
    const float* __restrict__ h, const float* __restrict__ agg,
    const _Float16* __restrict__ WtT,   // [D][D]
    const float* __restrict__ bt, float* __restrict__ out)
{
    int wave = threadIdx.x >> 5;
    int lane = threadIdx.x & 31;
    int g = lane >> 4;
    int n = lane & 15;
    int tile = blockIdx.x * 4 + wave;
    if (tile * 16 >= NN) return;
    int n0 = tile * 16;

    const float* hp = h   + (size_t)(n0 + n) * D;
    const float* ap = agg + (size_t)(n0 + n) * D;
    v16h A[4];
#pragma unroll
    for (int c = 0; c < 4; ++c) {
        v16h a;
#pragma unroll
        for (int j = 0; j < 8; ++j) {
            int kb = 32 * c + ((j < 4) ? (8 * g + 2 * j)
                                       : (16 + 8 * g + 2 * (j - 4)));
            float2 hv = *(const float2*)(hp + kb);
            float2 av = *(const float2*)(ap + kb);
            a[2 * j]     = (_Float16)siluf(hv.x + av.x);
            a[2 * j + 1] = (_Float16)siluf(hv.y + av.y);
        }
        A[c] = a;
    }

    for (int t = 0; t < 8; ++t) {
        int d = 16 * t + n;
        float bias = bt[d];
        v8f acc;
#pragma unroll
        for (int r = 0; r < 8; ++r) acc[r] = bias;
#pragma unroll
        for (int c = 0; c < 4; ++c) {
            const _Float16* bp = WtT + (size_t)d * D + 32 * c + 8 * g;
            acc = wmma_step(A[c], load_b16(bp), acc);
        }
#pragma unroll
        for (int r = 0; r < 8; ++r) {
            out[(size_t)(n0 + 8 * g + r) * D + d] = acc[r];
        }
    }
}

// ---------------------------------------------------------------------------
// Launch. Inputs (setup_inputs order):
//  0 node_embeddings [N,D] f32   1 edge_embeddings [E,D] f32
//  2 edge_index_list [2,E] i32   3 ln_gamma [D]          4 ln_beta [D]
//  5 W1 [D,H]  6 b1 [H]  7 W2 [H,D]  8 b2 [D]  9 Wt [D,D]  10 bt [D]
// Workspace: h (25.6MB) | agg (25.6MB) | W1T (64KB) | W2T (64KB) | WtT (32KB)
// ---------------------------------------------------------------------------
extern "C" void kernel_launch(void* const* d_in, const int* in_sizes, int n_in,
                              void* d_out, int out_size, void* d_ws, size_t ws_size,
                              hipStream_t stream) {
    const float* node_emb = (const float*)d_in[0];
    const float* edge_emb = (const float*)d_in[1];
    const int*   eidx     = (const int*)d_in[2];
    const float* gamma    = (const float*)d_in[3];
    const float* beta     = (const float*)d_in[4];
    const float* W1       = (const float*)d_in[5];
    const float* b1       = (const float*)d_in[6];
    const float* W2       = (const float*)d_in[7];
    const float* b2       = (const float*)d_in[8];
    const float* Wt       = (const float*)d_in[9];
    const float* bt       = (const float*)d_in[10];

    char* ws = (char*)d_ws;
    float*    h    = (float*)ws;
    float*    agg  = (float*)(ws + (size_t)NN * D * sizeof(float));
    _Float16* W1T  = (_Float16*)(ws + 2 * (size_t)NN * D * sizeof(float));
    _Float16* W2T  = W1T + (size_t)HID * D;   // contiguous after W1T
    _Float16* WtT  = W2T + (size_t)D * HID;

    ln_zero_kernel<<<(NN + 7) / 8, 256, 0, stream>>>(node_emb, gamma, beta, h, agg);
    prep_weights_kernel<<<(D * HID + 255) / 256, 256, 0, stream>>>(W1, W2, Wt, W1T, W2T, WtT);

    int etiles = NE / 16;                       // 40000 (exact)
    edge_mlp_kernel<<<(etiles + 3) / 4, 128, 0, stream>>>(
        edge_emb, eidx, eidx + NE, h, W1T, b1, b2, agg);

    int ntiles = NN / 16;                       // 3125 (exact)
    node_out_kernel<<<(ntiles + 3) / 4, 128, 0, stream>>>(
        h, agg, WtT, bt, (float*)d_out);
}
